// EfficientMemorySiLU_73461120631366
// MI455X (gfx1250) — matrix operand order repaired and checked by reading, not compiled, not run on hardware
//
#include <hip/hip_runtime.h>

// ---------------- problem constants ----------------
#define M_ROWS 8192               // 4*2048 rows after reshape(-1, d)
#define D_COLS 2048
#define NTOT   (M_ROWS * D_COLS)  // 16777216
#define BINS   8192
#define RANK   16
#define POWER_ITERS 6

// output layout (floats), concatenated in reference return order
#define OUT_RESULT 0
#define OUT_THRESH (NTOT)
#define OUT_L      (OUT_THRESH + 1)
#define OUT_R      (OUT_L + (size_t)M_ROWS * RANK)
#define OUT_SCALE  (OUT_R + (size_t)RANK * D_COLS)

// workspace layout (byte offsets)
#define WS_SCALARS 0                                  // [0]=absmax bits [1]=thresh [2]=residmax bits
#define WS_HIST    64                                 // BINS u32
#define WS_XSUB    (64 * 1024)                        // ushort[NTOT] (bf16)
#define WS_G       (WS_XSUB + (size_t)NTOT * 2)       // float[D*D]
#define WS_Q       (WS_G + (size_t)D_COLS * D_COLS * 4)
#define WS_Q2      (WS_Q + (size_t)D_COLS * RANK * 4)
#define WS_QB      (WS_Q2 + (size_t)D_COLS * RANK * 4) // ushort[D*RANK] (bf16)

typedef __attribute__((ext_vector_type(16))) __bf16 v16bf;
typedef __attribute__((ext_vector_type(8)))  float  v8f;

union BF16Frag {
    v16bf v;
    unsigned short u[16];
    unsigned int   w[8];
};

__device__ __forceinline__ unsigned short f2bf(float f) {
    unsigned int u = __float_as_uint(f);
    unsigned int rnd = 0x7FFFu + ((u >> 16) & 1u);
    return (unsigned short)((u + rnd) >> 16);
}

// ---------------- K0: zero scalars + histogram ----------------
__global__ void k_init(unsigned int* w, int n) {
    int i = blockIdx.x * blockDim.x + threadIdx.x;
    if (i < n) w[i] = 0u;
}

// ---------------- K1: SiLU + global abs-max ----------------
__global__ void __launch_bounds__(256)
k_silu_absmax(const float4* __restrict__ x, float4* __restrict__ out,
              unsigned int* __restrict__ absmax_bits, int n4) {
    __shared__ float red[256];
    int i = blockIdx.x * blockDim.x + threadIdx.x;
    float m = 0.f;
    if (i < n4) {
        float4 v = x[i];
        float4 r;
        r.x = v.x / (1.f + __expf(-v.x));
        r.y = v.y / (1.f + __expf(-v.y));
        r.z = v.z / (1.f + __expf(-v.z));
        r.w = v.w / (1.f + __expf(-v.w));
        out[i] = r;
        m = fmaxf(fmaxf(fabsf(v.x), fabsf(v.y)), fmaxf(fabsf(v.z), fabsf(v.w)));
    }
    red[threadIdx.x] = m;
    __syncthreads();
    for (int off = 128; off > 0; off >>= 1) {
        if (threadIdx.x < off) red[threadIdx.x] = fmaxf(red[threadIdx.x], red[threadIdx.x + off]);
        __syncthreads();
    }
    if (threadIdx.x == 0) atomicMax(absmax_bits, __float_as_uint(red[0]));
}

// ---------------- K2: |x| histogram (LDS-privatized) ----------------
__global__ void __launch_bounds__(256)
k_hist(const float* __restrict__ x, const unsigned int* __restrict__ scal,
       unsigned int* __restrict__ hist, int n) {
    __shared__ unsigned int h[BINS];
    for (int b = threadIdx.x; b < BINS; b += blockDim.x) h[b] = 0u;
    __syncthreads();
    float maxv = __uint_as_float(scal[0]);
    float inv = (maxv > 0.f) ? ((float)BINS / maxv) : 0.f;
    for (int i = blockIdx.x * blockDim.x + threadIdx.x; i < n; i += gridDim.x * blockDim.x) {
        float a = fabsf(x[i]);
        int b = (int)(a * inv);
        b = b < (BINS - 1) ? b : (BINS - 1);
        atomicAdd(&h[b], 1u);
    }
    __syncthreads();
    for (int b = threadIdx.x; b < BINS; b += blockDim.x)
        if (h[b]) atomicAdd(&hist[b], h[b]);
}

// ---------------- K3: select 0.99 quantile from histogram ----------------
__global__ void k_select(const unsigned int* __restrict__ hist, unsigned int* scal,
                         float* out_thresh, int n) {
    if (threadIdx.x == 0 && blockIdx.x == 0) {
        float maxv = __uint_as_float(scal[0]);
        unsigned long long target = (unsigned long long)(0.99 * (double)n);
        unsigned long long cum = 0;
        int b = 0;
        for (; b < BINS; ++b) { cum += hist[b]; if (cum >= target) break; }
        float thresh = maxv * ((float)(b + 1) / (float)BINS);
        scal[1] = __float_as_uint(thresh);
        *out_thresh = thresh;
    }
}

// ---------------- K4: xsub = clip-outliers(x) -> bf16 ----------------
__global__ void __launch_bounds__(256)
k_xsub(const float* __restrict__ x, const unsigned int* __restrict__ scal,
       unsigned short* __restrict__ xsub, int n) {
    int i = blockIdx.x * blockDim.x + threadIdx.x;
    if (i >= n) return;
    float thresh = __uint_as_float(scal[1]);
    float v = x[i];
    if (fabsf(v) > thresh) v = 0.f;
    xsub[i] = f2bf(v);
}

// ---------------- K5: Gram matrix G = Xsub^T * Xsub via WMMA bf16 ----------------
// block = 256 thr (8 waves), computes a 64x64 tile of G.
// wave w: row stripe wi = (w>>1)*16, column pair at wc = (w&1)*32 -> two 16x16
// subtiles sharing one A fragment (2 WMMAs per wave per k-step).
// Operands staged transposed in LDS so each half-fragment is a contiguous
// 16-byte read (compiler emits ds_load_b128 pairs).
__global__ void __launch_bounds__(256)
k_gram(const unsigned short* __restrict__ xs, float* __restrict__ G) {
    const int i0 = blockIdx.y * 64;
    const int j0 = blockIdx.x * 64;
    __shared__ unsigned short As[64][33];  // As[col_i][m]
    __shared__ unsigned short Bs[64][33];  // Bs[col_j][m]
    const int tid  = threadIdx.x;
    const int wave = tid >> 5;
    const int lane = tid & 31;
    const int wi = (wave >> 1) * 16;       // 0,16,32,48
    const int wj = (wave & 1) * 32;        // 0 or 32
    const int l15   = lane & 15;
    const int kbase = (lane < 16) ? 0 : 8;   // A: lanes16-31 hold K 8-15 / 24-31
    const int kb2   = (lane < 16) ? 0 : 8;   // B: lanes16-31 hold K 8-15 / 24-31
    v8f acc0 = {};
    v8f acc1 = {};
    for (int m0 = 0; m0 < M_ROWS; m0 += 32) {
        // stage 64 cols x 32 rows for each operand (2048 elems each, 8/thread)
        for (int e = 0; e < 8; ++e) {
            int idx = e * 256 + tid;          // 0..2047
            int mm = idx >> 6;                // 0..31
            int c  = idx & 63;                // 0..63
            size_t row = (size_t)(m0 + mm) * D_COLS;
            As[c][mm] = xs[row + i0 + c];
            Bs[c][mm] = xs[row + j0 + c];
        }
        // prefetch the chunk 2 iterations ahead (global_prefetch_b8)
        if (m0 + 64 < M_ROWS) {
            int mm = tid >> 6, c = tid & 63;
            size_t row = (size_t)(m0 + 64 + mm) * D_COLS;
            __builtin_prefetch(&xs[row + i0 + c], 0, 0);
            __builtin_prefetch(&xs[row + j0 + c], 0, 0);
        }
        __syncthreads();
        BF16Frag a, b0, b1;
        {   // A 16x32 bf16 fragment (ISA layout)
            int arow = wi + l15;
            for (int p = 0; p < 4; ++p) {
                int k = kbase + 2 * p;
                a.u[2 * p]     = As[arow][k];
                a.u[2 * p + 1] = As[arow][k + 1];
                int k2 = 16 + kbase + 2 * p;
                a.u[8 + 2 * p]     = As[arow][k2];
                a.u[8 + 2 * p + 1] = As[arow][k2 + 1];
            }
            // two B 32x16 bf16 fragments (cols wj.. and wj+16..)
            int bcol0 = wj + l15;
            int bcol1 = wj + 16 + l15;
            for (int p = 0; p < 8; ++p) {
                b0.u[2 * p]     = Bs[bcol0][kb2 + 2 * p + (p & 4) * 2];
                b0.u[2 * p + 1] = Bs[bcol0][kb2 + 2 * p + (p & 4) * 2 + 1];
                b1.u[2 * p]     = Bs[bcol1][kb2 + 2 * p + (p & 4) * 2];
                b1.u[2 * p + 1] = Bs[bcol1][kb2 + 2 * p + (p & 4) * 2 + 1];
            }
        }
        acc0 = __builtin_amdgcn_wmma_f32_16x16x32_bf16(false, a.v, false, b0.v,
                                                       (short)0, acc0, false, false);
        acc1 = __builtin_amdgcn_wmma_f32_16x16x32_bf16(false, a.v, false, b1.v,
                                                       (short)0, acc1, false, false);
        __syncthreads();
    }
    // C/D layout: N = lane%16; lanes16-31 carry M+8; VGPR r = M row
    int rbase = (lane < 16) ? 0 : 8;
    for (int r = 0; r < 8; ++r) {
        size_t grow = (size_t)(i0 + wi + rbase + r) * D_COLS;
        G[grow + (j0 + wj + l15)]      = acc0[r];
        G[grow + (j0 + wj + 16 + l15)] = acc1[r];
    }
}

// ---------------- K6: deterministic pseudo-random Q init ----------------
__global__ void k_qinit(float* Q) {
    int i = blockIdx.x * blockDim.x + threadIdx.x;
    if (i < D_COLS * RANK) {
        unsigned int h = (unsigned int)i * 2654435761u;
        h ^= h >> 16; h *= 2246822519u; h ^= h >> 13;
        Q[i] = ((float)(h & 0xFFFFu) / 32768.f) - 1.f;
    }
}

// ---------------- K7: Q2 = G @ Q (2048x2048 * 2048x16) ----------------
__global__ void __launch_bounds__(256)
k_gq(const float* __restrict__ G, const float* __restrict__ Q, float* __restrict__ Q2) {
    int idx = blockIdx.x * blockDim.x + threadIdx.x;
    if (idx >= D_COLS * RANK) return;
    int i = idx / RANK, k = idx % RANK;
    const float* g = G + (size_t)i * D_COLS;
    float acc = 0.f;
    for (int j = 0; j < D_COLS; ++j) acc += g[j] * Q[j * RANK + k];
    Q2[idx] = acc;
}

// ---------------- K8: modified Gram-Schmidt: Q = orth(Q2) ----------------
__global__ void __launch_bounds__(256) k_orth(float* Q2, float* Q) {
    __shared__ float red[256];
    const int tid = threadIdx.x;
    for (int k = 0; k < RANK; ++k) {
        for (int j = 0; j < k; ++j) {
            float d = 0.f;
            for (int i = tid; i < D_COLS; i += 256) d += Q2[i * RANK + k] * Q[i * RANK + j];
            red[tid] = d; __syncthreads();
            for (int off = 128; off; off >>= 1) {
                if (tid < off) red[tid] += red[tid + off];
                __syncthreads();
            }
            float dot = red[0]; __syncthreads();
            for (int i = tid; i < D_COLS; i += 256) Q2[i * RANK + k] -= dot * Q[i * RANK + j];
            __syncthreads();
        }
        float d = 0.f;
        for (int i = tid; i < D_COLS; i += 256) { float v = Q2[i * RANK + k]; d += v * v; }
        red[tid] = d; __syncthreads();
        for (int off = 128; off; off >>= 1) {
            if (tid < off) red[tid] += red[tid + off];
            __syncthreads();
        }
        float inv = rsqrtf(fmaxf(red[0], 1e-20f)); __syncthreads();
        for (int i = tid; i < D_COLS; i += 256) Q[i * RANK + k] = Q2[i * RANK + k] * inv;
        __syncthreads();
    }
}

// ---------------- K9: R = Q^T (to output) and Qb = bf16(Q) ----------------
__global__ void k_writeR(const float* __restrict__ Q, float* __restrict__ Rout,
                         unsigned short* __restrict__ Qb) {
    int idx = blockIdx.x * blockDim.x + threadIdx.x;
    if (idx >= D_COLS * RANK) return;
    int i = idx / RANK, k = idx % RANK;
    float v = Q[idx];
    Rout[(size_t)k * D_COLS + i] = v;
    Qb[idx] = f2bf(v);
}

// ---------------- K10: L = Xsub @ Q via WMMA bf16 (8192x2048 * 2048x16) ----------------
// block = 256 thr (8 waves); wave w does rows [m0+16w, m0+16w+16), all 16 cols
__global__ void __launch_bounds__(256)
k_lmm(const unsigned short* __restrict__ xs, const unsigned short* __restrict__ qb,
      float* __restrict__ Lout) {
    const int wave = threadIdx.x >> 5;
    const int lane = threadIdx.x & 31;
    const int m0 = blockIdx.x * 128 + wave * 16;
    v8f acc = {};
    const int arow  = m0 + (lane & 15);
    const int kbase = (lane < 16) ? 0 : 8;
    const int kb2   = (lane < 16) ? 0 : 8;
    const int col   = lane & 15;
    for (int d0 = 0; d0 < D_COLS; d0 += 32) {
        BF16Frag a, b;
        // A rows are row-major in memory: K pairs are contiguous -> direct 32-bit loads
        const unsigned int* xrow = (const unsigned int*)(xs + (size_t)arow * D_COLS + d0);
        for (int p = 0; p < 4; ++p) {
            a.w[p]     = xrow[(kbase >> 1) + p];          // K = kbase..kbase+7
            a.w[4 + p] = xrow[((16 + kbase) >> 1) + p];   // K = 16+kbase..+7
        }
        // B (Qb) is K-major 2048x16; small & L2-hot, gather directly
        for (int p = 0; p < 8; ++p) {
            b.u[2 * p]     = qb[(size_t)(d0 + kb2 + 2 * p + (p & 4) * 2) * RANK + col];
            b.u[2 * p + 1] = qb[(size_t)(d0 + kb2 + 2 * p + (p & 4) * 2 + 1) * RANK + col];
        }
        acc = __builtin_amdgcn_wmma_f32_16x16x32_bf16(false, a.v, false, b.v,
                                                      (short)0, acc, false, false);
    }
    int rbase = (lane < 16) ? 0 : 8;
    for (int r = 0; r < 8; ++r)
        Lout[(size_t)(m0 + rbase + r) * RANK + col] = acc[r];
}

// ---------------- K11: residual max |xsub - L@R| ----------------
__global__ void __launch_bounds__(256)
k_resid(const float* __restrict__ x, const unsigned int* __restrict__ scal,
        const float* __restrict__ Lf, const float* __restrict__ Rf,
        unsigned int* __restrict__ residmax_bits) {
    __shared__ float lrow[RANK];
    __shared__ float red[256];
    const int m = blockIdx.x;
    const float thresh = __uint_as_float(scal[1]);
    if (threadIdx.x < RANK) lrow[threadIdx.x] = Lf[(size_t)m * RANK + threadIdx.x];
    __syncthreads();
    float mx = 0.f;
    for (int d = threadIdx.x; d < D_COLS; d += 256) {
        float v = x[(size_t)m * D_COLS + d];
        if (fabsf(v) > thresh) v = 0.f;
        float acc = 0.f;
#pragma unroll
        for (int k = 0; k < RANK; ++k) acc += lrow[k] * Rf[(size_t)k * D_COLS + d];
        mx = fmaxf(mx, fabsf(v - acc));
    }
    red[threadIdx.x] = mx; __syncthreads();
    for (int off = 128; off; off >>= 1) {
        if (threadIdx.x < off) red[threadIdx.x] = fmaxf(red[threadIdx.x], red[threadIdx.x + off]);
        __syncthreads();
    }
    if (threadIdx.x == 0) atomicMax(residmax_bits, __float_as_uint(red[0]));
}

// ---------------- K12: scale = residmax / 127 ----------------
__global__ void k_final(const unsigned int* __restrict__ scal, float* out_scale) {
    if (threadIdx.x == 0 && blockIdx.x == 0)
        *out_scale = __uint_as_float(scal[2]) / 127.0f;
}

// ---------------- host launcher ----------------
extern "C" void kernel_launch(void* const* d_in, const int* in_sizes, int n_in,
                              void* d_out, int out_size, void* d_ws, size_t ws_size,
                              hipStream_t stream) {
    const float* x = (const float*)d_in[0];
    float* out = (float*)d_out;
    char* ws = (char*)d_ws;

    unsigned int*   scal = (unsigned int*)(ws + WS_SCALARS);
    unsigned int*   hist = (unsigned int*)(ws + WS_HIST);
    unsigned short* xsub = (unsigned short*)(ws + WS_XSUB);
    float*          G    = (float*)(ws + WS_G);
    float*          Q    = (float*)(ws + WS_Q);
    float*          Q2   = (float*)(ws + WS_Q2);
    unsigned short* Qb   = (unsigned short*)(ws + WS_QB);

    float* out_result = out + OUT_RESULT;
    float* out_thresh = out + OUT_THRESH;
    float* out_L      = out + OUT_L;
    float* out_R      = out + OUT_R;
    float* out_scale  = out + OUT_SCALE;

    // K0: zero scalars + hist (every call: atomics accumulate)
    {
        int n = BINS + 16;
        k_init<<<(n + 255) / 256, 256, 0, stream>>>(scal, n);
    }
    // K1: SiLU + absmax
    {
        int n4 = NTOT / 4;
        k_silu_absmax<<<n4 / 256, 256, 0, stream>>>((const float4*)x, (float4*)out_result,
                                                    scal + 0, n4);
    }
    // K2: histogram
    k_hist<<<2048, 256, 0, stream>>>(x, scal, hist, NTOT);
    // K3: threshold selection
    k_select<<<1, 1, 0, stream>>>(hist, scal, out_thresh, NTOT);
    // K4: clipped bf16 matrix
    k_xsub<<<NTOT / 256, 256, 0, stream>>>(x, scal, xsub, NTOT);
    // K5: Gram matrix (WMMA, 64x64 tiles)
    {
        dim3 grid(D_COLS / 64, D_COLS / 64);
        k_gram<<<grid, 256, 0, stream>>>(xsub, G);
    }
    // K6 + power iteration (K7/K8)
    k_qinit<<<(D_COLS * RANK + 255) / 256, 256, 0, stream>>>(Q);
    for (int it = 0; it < POWER_ITERS; ++it) {
        k_gq<<<(D_COLS * RANK + 255) / 256, 256, 0, stream>>>(G, Q, Q2);
        k_orth<<<1, 256, 0, stream>>>(Q2, Q);
    }
    // K9: R = Q^T, Qb = bf16(Q)
    k_writeR<<<(D_COLS * RANK + 255) / 256, 256, 0, stream>>>(Q, out_R, Qb);
    // K10: L = Xsub @ Q (WMMA)
    k_lmm<<<M_ROWS / 128, 256, 0, stream>>>(xsub, Qb, out_L);
    // K11: residual abs-max
    k_resid<<<M_ROWS, 256, 0, stream>>>(x, scal, out_L, out_R, scal + 2);
    // K12: scale
    k_final<<<1, 1, 0, stream>>>(scal, out_scale);
}
